// Block1_32787780337772
// MI455X (gfx1250) — matrix operand order repaired and checked by reading, not compiled, not run on hardware
//
#include <hip/hip_runtime.h>
#include <hip/hip_bf16.h>

typedef __attribute__((ext_vector_type(16))) _Float16 v16h;
typedef __attribute__((ext_vector_type(8)))  _Float16 v8h;
typedef __attribute__((ext_vector_type(8)))  float    v8f;

__device__ __forceinline__ v16h cat16(v8h lo, v8h hi) {
    return __builtin_shufflevector(lo, hi, 0, 1, 2, 3, 4, 5, 6, 7,
                                   8, 9, 10, 11, 12, 13, 14, 15);
}

// ---------------------------------------------------------------------------
// Weight prep: f32 -> f16 copy, and 3x3-conv weight reorder OIHW -> [tap][O][I]
// ---------------------------------------------------------------------------
__global__ __launch_bounds__(256) void f32_to_f16_kernel(
    const float* __restrict__ s, _Float16* __restrict__ d, int n)
{
    int i = blockIdx.x * blockDim.x + threadIdx.x;
    if (i < n) d[i] = (_Float16)s[i];
}

__global__ __launch_bounds__(256) void convw_reorder_kernel(
    const float* __restrict__ src, _Float16* __restrict__ dst)
{
    int idx = blockIdx.x * blockDim.x + threadIdx.x;   // [tap][o][c], 9*64*128
    if (idx >= 9 * 64 * 128) return;
    int c = idx & 127;
    int o = (idx >> 7) & 63;
    int t = idx >> 13;                                  // ky*3+kx
    int ky = t / 3, kx = t - 3 * ky;
    dst[idx] = (_Float16)src[(((size_t)o * 128 + c) * 3 + ky) * 3 + kx];
}

// ---------------------------------------------------------------------------
// LayerNorm over channels; input NCHW (d_in) or position-major; output is
// always position-major f16 [Ntot, C] (contiguous channel rows for WMMA A).
// ---------------------------------------------------------------------------
__global__ __launch_bounds__(256) void lnorm_kernel(
    const float* __restrict__ X, const float* __restrict__ g,
    const float* __restrict__ bt, _Float16* __restrict__ Y,
    int C, int HW, int Btot, int inNCHW)
{
    int n = blockIdx.x * blockDim.x + threadIdx.x;
    if (n >= Btot * HW) return;
    int b = n / HW, p = n - b * HW;
    const float* xp = inNCHW ? (X + (size_t)b * C * HW + p) : (X + (size_t)n * C);
    size_t cstep = inNCHW ? (size_t)HW : (size_t)1;
    float s = 0.f, s2 = 0.f;
    for (int c = 0; c < C; ++c) {
        float v = xp[(size_t)c * cstep];
        s += v; s2 += v * v;
    }
    float mu  = s / (float)C;
    float var = s2 / (float)C - mu * mu;
    float inv = rsqrtf(var + 1e-5f);
    _Float16* yp = Y + (size_t)n * C;
    for (int c = 0; c < C; ++c) {
        float v = xp[(size_t)c * cstep];
        yp[c] = (_Float16)((v - mu) * inv * g[c] + bt[c]);
    }
}

// ---------------------------------------------------------------------------
// Position-major GEMM: Y[n, oc] = sum_c X[n,c] * W[oc,c] + bias[oc]
// A = activations (16 pos x 32 ch): two contiguous b128 loads per lane.
// B = weights    (32 ch x 16 oc):  two contiguous b128 loads per lane.
// One wave -> 16 pos x 64 oc (4 accumulators share the A fragment).
// ---------------------------------------------------------------------------
__global__ __launch_bounds__(256) void gemm_wmma_f16(
    const _Float16* __restrict__ Wh, const float* __restrict__ bias,
    const _Float16* __restrict__ X, int Ci, int Co, int HW, int Btot,
    float* __restrict__ Yf, _Float16* __restrict__ Yh,
    int outCtot, int outCoff, int outNCHW,
    const float* __restrict__ Res, int resCtot, int resNCHW, int gelu)
{
    const int lane = threadIdx.x & 31;
    const int wave = (blockIdx.x * blockDim.x + threadIdx.x) >> 5;
    const int nGroups = Co >> 6;                        // 64-oc groups
    const int pTiles  = (Btot * HW) >> 4;               // 16-position tiles
    if (wave >= nGroups * pTiles) return;               // wave-uniform
    const int g   = wave % nGroups;
    const int pt  = wave / nGroups;
    const int n0  = pt << 4;
    const int ocg = g << 6;

    const int hi   = lane >> 4;                         // 0 / 1
    const int mrow = lane & 15;                         // A: M (position)
    const int ncol = lane & 15;                         // B/D: N (out ch)
    const int kb   = hi << 4;                           // B: K base 0 / 16

    const _Float16* xr = X + (size_t)(n0 + mrow) * Ci;
    v8f acc[4] = {{}, {}, {}, {}};
    for (int k0 = 0; k0 < Ci; k0 += 32) {
        if (k0 + 32 < Ci)                               // -> global_prefetch_b8
            __builtin_prefetch(xr + k0 + 32, 0, 0);
        v16h a = cat16(*(const v8h*)(xr + k0 + (hi << 4)),
                       *(const v8h*)(xr + k0 + 8 + (hi << 4)));
#pragma unroll
        for (int t = 0; t < 4; ++t) {
            const _Float16* wrb = Wh + (size_t)(ocg + t * 16 + ncol) * Ci + k0 + kb;
            v16h bm = cat16(*(const v8h*)wrb, *(const v8h*)(wrb + 8));
            acc[t] = __builtin_amdgcn_wmma_f32_16x16x32_f16(
                false, a, false, bm, (short)0, acc[t], false, false);
        }
    }
    const int b  = n0 / HW;
    const int p0 = n0 - b * HW;
#pragma unroll
    for (int t = 0; t < 4; ++t) {
        int oc = ocg + t * 16 + ncol;
        float bi = bias[oc];
#pragma unroll
        for (int r = 0; r < 8; ++r) {
            int p = p0 + r + (hi << 3);                 // D: M = r + 8*hi
            int n = n0 + r + (hi << 3);
            float v = acc[t][r] + bi;
            if (Res) {
                size_t ri = resNCHW ? (((size_t)b * resCtot + oc) * HW + p)
                                    : ((size_t)n * resCtot + oc);
                v += Res[ri];
            }
            if (gelu) v = 0.5f * v * (1.0f + erff(v * 0.70710678118654752f));
            if (Yf) {
                size_t oi = outNCHW ? (((size_t)b * outCtot + outCoff + oc) * HW + p)
                                    : ((size_t)n * outCtot + outCoff + oc);
                Yf[oi] = v;
            }
            if (Yh) Yh[(size_t)n * outCtot + outCoff + oc] = (_Float16)v;
        }
    }
}

// ---------------------------------------------------------------------------
// 3x3 conv (pad 1), implicit GEMM, position-major: A = shifted activation
// rows (zero-filled at image borders, checked per lane), B = pre-reordered
// [tap][o][c] f16 weights. One wave -> 16 pos x 64 oc, 36 k-steps.
// ---------------------------------------------------------------------------
__global__ __launch_bounds__(256) void conv3x3_wmma(
    const _Float16* __restrict__ Wh, const float* __restrict__ bias,
    const _Float16* __restrict__ X, float* __restrict__ Y,
    int H, int W, int Btot)
{
    const int Ci = 128;
    const int lane = threadIdx.x & 31;
    const int wave = (blockIdx.x * blockDim.x + threadIdx.x) >> 5;
    const int HWl = H * W;
    const int pTiles = (Btot * HWl) >> 4;
    if (wave >= pTiles) return;
    const int n0 = wave << 4;
    const int b  = n0 / HWl;
    const int p0 = n0 - b * HWl;
    const int h  = p0 / W, w0 = p0 - h * W;             // 16 | W: row-contained
    const int hi = lane >> 4, mrow = lane & 15, ncol = lane & 15, kb = hi << 4;
    const int wm = w0 + mrow;                           // this lane's column

    const v8h zero8 = {};
    v8f acc[4] = {{}, {}, {}, {}};
    for (int ky = 0; ky < 3; ++ky) {
        const int  hh  = h + ky - 1;
        const bool rok = (hh >= 0) && (hh < H);
        for (int kx = 0; kx < 3; ++kx) {
            const int  wwm = wm + kx - 1;
            const bool aok = rok && (wwm >= 0) && (wwm < W);
            const _Float16* ar = X + (size_t)(b * HWl + hh * W + wwm) * Ci;
            const _Float16* wt = Wh + (size_t)(ky * 3 + kx) * 64 * 128;
            for (int ct = 0; ct < Ci; ct += 32) {
                v8h alo = aok ? *(const v8h*)(ar + ct + (hi << 4)) : zero8;
                v8h ahi = aok ? *(const v8h*)(ar + ct + 8 + (hi << 4)) : zero8;
                v16h a = cat16(alo, ahi);
#pragma unroll
                for (int t = 0; t < 4; ++t) {
                    const _Float16* wrb = wt + (size_t)(t * 16 + ncol) * 128 + ct + kb;
                    v16h bm = cat16(*(const v8h*)wrb, *(const v8h*)(wrb + 8));
                    acc[t] = __builtin_amdgcn_wmma_f32_16x16x32_f16(
                        false, a, false, bm, (short)0, acc[t], false, false);
                }
            }
        }
    }
#pragma unroll
    for (int t = 0; t < 4; ++t) {
#pragma unroll
        for (int r = 0; r < 8; ++r) {
            int oc = t * 16 + ncol;
            Y[(size_t)(n0 + r + (hi << 3)) * 64 + oc] = acc[t][r] + bias[oc];
        }
    }
}

// ---------------------------------------------------------------------------
// Axial attention (dh=8), position-major q/k/v: one block = (b, fixed, head),
// 128 threads (thread i = query position). 8 contiguous channels per row.
// ---------------------------------------------------------------------------
__global__ __launch_bounds__(128) void axial_attn_kernel(
    const float* __restrict__ Q, const float* __restrict__ K,
    const float* __restrict__ V,
    _Float16* __restrict__ Otmp, float* __restrict__ Amap,
    int Btot, int Cc, int H, int W, int heightAxial, float scale)
{
    const int heads = 8, dh = 8;
    const int L = heightAxial ? H : W;
    const int F = heightAxial ? W : H;
    int blk  = blockIdx.x;
    int head = blk % heads;
    int f    = (blk / heads) % F;
    int b    = blk / (heads * F);
    int i    = threadIdx.x;
    int hc   = head * dh;

    __shared__ float kk[128 * 8];       // [j][d]
    __shared__ float vv[128 * 8];

    auto npos = [&](int j) {
        return b * H * W + (heightAxial ? j * W + f : f * W + j);
    };
    {
        int j = threadIdx.x;
        const float* kp = K + (size_t)npos(j) * Cc + hc;
        const float* vp = V + (size_t)npos(j) * Cc + hc;
#pragma unroll
        for (int d = 0; d < dh; ++d) { kk[j * 8 + d] = kp[d]; vv[j * 8 + d] = vp[d]; }
    }
    __syncthreads();

    const float* qp = Q + (size_t)npos(i) * Cc + hc;
    float qr[8];
#pragma unroll
    for (int d = 0; d < dh; ++d) qr[d] = qp[d];

    float m = -3.0e38f;
    for (int j = 0; j < L; ++j) {
        float s = 0.f;
#pragma unroll
        for (int d = 0; d < dh; ++d) s += qr[d] * kk[j * 8 + d];
        m = fmaxf(m, s * scale);
    }
    float sum = 0.f;
    float acc[8] = {0.f, 0.f, 0.f, 0.f, 0.f, 0.f, 0.f, 0.f};
    for (int j = 0; j < L; ++j) {
        float s = 0.f;
#pragma unroll
        for (int d = 0; d < dh; ++d) s += qr[d] * kk[j * 8 + d];
        float e = expf(s * scale - m);
        sum += e;
#pragma unroll
        for (int d = 0; d < dh; ++d) acc[d] += e * vv[j * 8 + d];
    }
    float inv = 1.0f / sum;
    float* arow = Amap + ((((size_t)b * F + f) * heads + head) * L + i) * (size_t)L;
    for (int j = 0; j < L; ++j) {
        float s = 0.f;
#pragma unroll
        for (int d = 0; d < dh; ++d) s += qr[d] * kk[j * 8 + d];
        arow[j] = expf(s * scale - m) * inv;
    }
    _Float16* op = Otmp + (size_t)npos(i) * Cc + hc;
#pragma unroll
    for (int d = 0; d < dh; ++d) op[d] = (_Float16)(acc[d] * inv);
}

// ---------------------------------------------------------------------------
// Fused depthwise 1/3/5/7 convs + GELU, position-major in/out.
// Consecutive threads = consecutive channels -> fully coalesced taps.
// ---------------------------------------------------------------------------
__global__ __launch_bounds__(256) void dwmix_kernel(
    const float* __restrict__ X,            // [Ntot, 256]
    const float* __restrict__ dw1, const float* __restrict__ db1,
    const float* __restrict__ dw2, const float* __restrict__ db2,
    const float* __restrict__ dw3, const float* __restrict__ db3,
    const float* __restrict__ dw4, const float* __restrict__ db4,
    _Float16* __restrict__ Out,             // [Ntot, 1024]
    int H, int W, int Btot)
{
    const int hid = 256;
    int idx = blockIdx.x * blockDim.x + threadIdx.x;
    if (idx >= Btot * H * W * hid) return;
    int ch = idx & (hid - 1);
    int n  = idx >> 8;
    int w  = n % W;
    int h  = (n / W) % H;
    int b  = n / (W * H);
    float s1 = db1[ch], s2 = db2[ch], s3 = db3[ch], s4 = db4[ch];
    for (int dy = -3; dy <= 3; ++dy) {
        int yy = h + dy;
        for (int dx = -3; dx <= 3; ++dx) {
            int xx = w + dx;
            float v = 0.0f;
            if (yy >= 0 && yy < H && xx >= 0 && xx < W)
                v = X[(size_t)(b * H * W + yy * W + xx) * hid + ch];
            s4 += v * dw4[ch * 49 + (dy + 3) * 7 + (dx + 3)];
            if (dy >= -2 && dy <= 2 && dx >= -2 && dx <= 2)
                s3 += v * dw3[ch * 25 + (dy + 2) * 5 + (dx + 2)];
            if (dy >= -1 && dy <= 1 && dx >= -1 && dx <= 1)
                s2 += v * dw2[ch * 9 + (dy + 1) * 3 + (dx + 1)];
            if (dy == 0 && dx == 0)
                s1 += v * dw1[ch];
        }
    }
    auto gelu = [](float x) { return 0.5f * x * (1.0f + erff(x * 0.70710678118654752f)); };
    _Float16* op = Out + (size_t)n * 1024 + ch;
    op[0]   = (_Float16)gelu(s1);
    op[256] = (_Float16)gelu(s2);
    op[512] = (_Float16)gelu(s3);
    op[768] = (_Float16)gelu(s4);
}

// ---------------------------------------------------------------------------
extern "C" void kernel_launch(void* const* d_in, const int* in_sizes, int n_in,
                              void* d_out, int out_size, void* d_ws, size_t ws_size,
                              hipStream_t stream) {
    (void)in_sizes; (void)n_in; (void)out_size; (void)ws_size;
    const int Bn = 4, Cc = 64, Hh = 128, Ww = 128;
    const int HW = Hh * Ww;                        // 16384
    const int Ntot = Bn * HW;                      // 65536
    const float scale = 0.125f;                    // 64^-0.5

    // ---- inputs (setup_inputs dict order, params flattened depth-first) ----
    const float* pan  = (const float*)d_in[0];
    const float* lms  = (const float*)d_in[1];
    const float* ln1g = (const float*)d_in[2];  const float* ln1b = (const float*)d_in[3];
    const float* ln2g = (const float*)d_in[4];  const float* ln2b = (const float*)d_in[5];
    const float* ln3g = (const float*)d_in[6];  const float* ln3b = (const float*)d_in[7];
    const float* h_wq = (const float*)d_in[8];  const float* h_bq = (const float*)d_in[9];
    const float* h_wk = (const float*)d_in[10]; const float* h_bk = (const float*)d_in[11];
    const float* h_wv = (const float*)d_in[12]; const float* h_bv = (const float*)d_in[13];
    const float* h_wo = (const float*)d_in[14]; const float* h_bo = (const float*)d_in[15];
    const float* w_wq = (const float*)d_in[16]; const float* w_bq = (const float*)d_in[17];
    const float* w_wk = (const float*)d_in[18]; const float* w_bk = (const float*)d_in[19];
    const float* w_wv = (const float*)d_in[20]; const float* w_bv = (const float*)d_in[21];
    const float* w_wo = (const float*)d_in[22]; const float* w_bo = (const float*)d_in[23];
    const float* convw = (const float*)d_in[24]; const float* convb = (const float*)d_in[25];
    const float* m_w1  = (const float*)d_in[26]; const float* m_b1  = (const float*)d_in[27];
    const float* m_dw1 = (const float*)d_in[28]; const float* m_db1 = (const float*)d_in[29];
    const float* m_dw2 = (const float*)d_in[30]; const float* m_db2 = (const float*)d_in[31];
    const float* m_dw3 = (const float*)d_in[32]; const float* m_db3 = (const float*)d_in[33];
    const float* m_dw4 = (const float*)d_in[34]; const float* m_db4 = (const float*)d_in[35];
    const float* m_w3  = (const float*)d_in[36]; const float* m_b3  = (const float*)d_in[37];

    // ---- workspace carve-out (all activations position-major) ----
    char* ws = (char*)d_ws;
    size_t off = 0;
    auto take = [&](size_t bytes) -> char* {
        char* p = ws + off;
        off += (bytes + 255) & ~(size_t)255;
        return p;
    };
    _Float16* pan_h = (_Float16*)take((size_t)Ntot * 64 * 2);
    _Float16* lms_h = (_Float16*)take((size_t)Ntot * 64 * 2);
    float* qh = (float*)take((size_t)Ntot * 64 * 4);
    float* kh = (float*)take((size_t)Ntot * 64 * 4);
    float* vh = (float*)take((size_t)Ntot * 64 * 4);
    float* qw = (float*)take((size_t)Ntot * 64 * 4);
    float* kw = (float*)take((size_t)Ntot * 64 * 4);
    float* vw = (float*)take((size_t)Ntot * 64 * 4);
    _Float16* oth  = (_Float16*)take((size_t)Ntot * 64 * 2);
    _Float16* otw  = (_Float16*)take((size_t)Ntot * 64 * 2);
    _Float16* cat2 = (_Float16*)take((size_t)Ntot * 128 * 2);
    float*    y0   = (float*)take((size_t)Ntot * 64 * 4);
    _Float16* outn = (_Float16*)take((size_t)Ntot * 64 * 2);
    float*    h1   = (float*)take((size_t)Ntot * 256 * 4);
    _Float16* catg = (_Float16*)take((size_t)Ntot * 1024 * 2);
    // f16 weight arena
    _Float16* h_wq_h = (_Float16*)take(4096 * 2);
    _Float16* h_wk_h = (_Float16*)take(4096 * 2);
    _Float16* h_wv_h = (_Float16*)take(4096 * 2);
    _Float16* h_wo_h = (_Float16*)take(4096 * 2);
    _Float16* w_wq_h = (_Float16*)take(4096 * 2);
    _Float16* w_wk_h = (_Float16*)take(4096 * 2);
    _Float16* w_wv_h = (_Float16*)take(4096 * 2);
    _Float16* w_wo_h = (_Float16*)take(4096 * 2);
    _Float16* m_w1_h = (_Float16*)take(16384 * 2);
    _Float16* m_w3_h = (_Float16*)take(65536 * 2);
    _Float16* convw_h = (_Float16*)take(73728 * 2);   // [tap][o][c]

    // ---- d_out partition: [output(NCHW) | amap_h | amap_w] ----
    float* out0  = (float*)d_out;
    float* amapH = out0 + (size_t)Ntot * 64 / HW * HW;   // = Ntot*... (B*C*H*W)
    amapH = out0 + (size_t)Bn * 64 * HW;
    float* amapW = amapH + (size_t)Bn * Ww * 8 * Hh * Hh;

    auto cvt = [&](const float* s, _Float16* d, int n) {
        f32_to_f16_kernel<<<(n + 255) / 256, 256, 0, stream>>>(s, d, n);
    };
    auto gemm = [&](const _Float16* Wh, const float* bias, const _Float16* X,
                    int Ci, int Co, float* Yf, _Float16* Yh,
                    int outCtot, int outCoff, int outNCHW,
                    const float* Res, int resCtot, int resNCHW, int act) {
        int waves = (Co / 64) * (Ntot / 16);
        int blocks = (waves * 32) / 256;
        gemm_wmma_f16<<<blocks, 256, 0, stream>>>(
            Wh, bias, X, Ci, Co, HW, Bn, Yf, Yh,
            outCtot, outCoff, outNCHW, Res, resCtot, resNCHW, act);
    };

    // 0) weight prep (f16 conversions + conv reorder)
    cvt(h_wq, h_wq_h, 4096);  cvt(h_wk, h_wk_h, 4096);
    cvt(h_wv, h_wv_h, 4096);  cvt(h_wo, h_wo_h, 4096);
    cvt(w_wq, w_wq_h, 4096);  cvt(w_wk, w_wk_h, 4096);
    cvt(w_wv, w_wv_h, 4096);  cvt(w_wo, w_wo_h, 4096);
    cvt(m_w1, m_w1_h, 16384); cvt(m_w3, m_w3_h, 65536);
    convw_reorder_kernel<<<(9 * 64 * 128 + 255) / 256, 256, 0, stream>>>(convw, convw_h);

    // 1) LayerNorms (NCHW in -> position-major f16 out)
    lnorm_kernel<<<Ntot / 256, 256, 0, stream>>>(pan, ln1g, ln1b, pan_h, 64, HW, Bn, 1);
    lnorm_kernel<<<Ntot / 256, 256, 0, stream>>>(lms, ln2g, ln2b, lms_h, 64, HW, Bn, 1);

    // 2) q/k/v projections (position-major f32 outputs)
    gemm(h_wq_h, h_bq, pan_h, 64, 64, qh, nullptr, 64, 0, 0, nullptr, 64, 0, 0);
    gemm(h_wk_h, h_bk, lms_h, 64, 64, kh, nullptr, 64, 0, 0, nullptr, 64, 0, 0);
    gemm(h_wv_h, h_bv, lms_h, 64, 64, vh, nullptr, 64, 0, 0, nullptr, 64, 0, 0);
    gemm(w_wq_h, w_bq, pan_h, 64, 64, qw, nullptr, 64, 0, 0, nullptr, 64, 0, 0);
    gemm(w_wk_h, w_bk, lms_h, 64, 64, kw, nullptr, 64, 0, 0, nullptr, 64, 0, 0);
    gemm(w_wv_h, w_bv, lms_h, 64, 64, vw, nullptr, 64, 0, 0, nullptr, 64, 0, 0);

    // 3) axial attention; amaps straight into d_out
    axial_attn_kernel<<<Bn * Ww * 8, 128, 0, stream>>>(
        qh, kh, vh, oth, amapH, Bn, Cc, Hh, Ww, 1, scale);
    axial_attn_kernel<<<Bn * Hh * 8, 128, 0, stream>>>(
        qw, kw, vw, otw, amapW, Bn, Cc, Hh, Ww, 0, scale);

    // 4) o-projections + lms(NCHW) residual -> cat2 [Ntot,128] f16
    gemm(h_wo_h, h_bo, oth, 64, 64, nullptr, cat2, 128, 0,  0, lms, 64, 1, 0);
    gemm(w_wo_h, w_bo, otw, 64, 64, nullptr, cat2, 128, 64, 0, lms, 64, 1, 0);

    // 5) 3x3 conv (implicit GEMM) -> y0 [Ntot,64] f32
    conv3x3_wmma<<<((Ntot / 16) * 32) / 256, 256, 0, stream>>>(
        convw_h, convb, cat2, y0, Hh, Ww, Bn);

    // 6) LN3 (position-major in) -> outn f16
    lnorm_kernel<<<Ntot / 256, 256, 0, stream>>>(y0, ln3g, ln3b, outn, 64, HW, Bn, 0);

    // 7) MLP w1 (64->256) + GELU -> h1 [Ntot,256] f32
    gemm(m_w1_h, m_b1, outn, 64, 256, h1, nullptr, 256, 0, 0, nullptr, 64, 0, 1);

    // 8) depthwise 1/3/5/7 + GELU -> catg [Ntot,1024] f16
    dwmix_kernel<<<((size_t)Ntot * 256) / 256, 256, 0, stream>>>(
        h1, m_dw1, m_db1, m_dw2, m_db2, m_dw3, m_db3, m_dw4, m_db4, catg, Hh, Ww, Bn);

    // 9) MLP w3 (1024->64) + y0 residual -> final output (NCHW) in d_out
    gemm(m_w3_h, m_b3, catg, 1024, 64, out0, nullptr, 64, 0, 1, y0, 64, 0, 0);
}
